// TopoteinModelV0_747324310091
// MI455X (gfx1250) — compile-verified, matrix operand order checked by reading, not run on hardware
//
#include <hip/hip_runtime.h>

// ============================================================================
// Topotein GNN forward for MI455X (gfx1250, wave32, WMMA).
//
// Roofline reasoning (no runtime available):
//  * 6 layers dominated by dense binary-adjacency GEMMs (~148 GFLOP total,
//    ~3.4 GB/pass of f32 adjacency reads) -> memory bound at 23.3 TB/s HBM.
//  * Adjacency is {0,1} -> exact in bf16. One-time bf16 conversion halves the
//    dominant traffic; the per-layer adjacency working set (~288 MB) ~= the
//    192 MB L2, so layers 2..6 mostly stream from L2.
//  * Big GEMMs on v_wmma_f32_16x16x32_bf16 (f32 accum). A tiles staged with
//    GLOBAL_LOAD_ASYNC_TO_LDS_B128 + ASYNCcnt double buffering (CDNA5 async
//    path) so the adjacency stream overlaps the WMMAs; B tiles (tiny, L2-hot
//    feature matrices) are register-staged + transposed into [N][K] LDS.
//  * BN / residual / ReLU / segment-mean in f32; fully deterministic
//    (no atomics, fixed-order reductions).
//
// Workspace: ~320 MB (bf16 adjacency copies dominate).
// Param flattening assumed in setup_inputs() insertion order:
//   d_in[0..6]  = x, edge_attr, sse_attr, B0, B1, A1, batch_idx
//   d_in[7..12] = Win0.W,b, Win1.W,b, Win2.W,b
//   d_in[13+2*(fr*3+to)] = Wemb[fr][to].W, +1 = .b
//   d_in[31..36]= bn g0,b0,g1,b1,g2,b2
// ============================================================================

typedef __attribute__((ext_vector_type(8)))  float        v8f;
typedef __attribute__((ext_vector_type(16))) __bf16       v16bf;
typedef __attribute__((ext_vector_type(4)))  unsigned int v4u;
typedef __attribute__((ext_vector_type(8)))  unsigned int v8u;

// Exact pointee type of the async-LDS builtin's pointer params (per clang
// diagnostic: "__attribute__((__vector_size__(4 * sizeof(int)))) int *").
typedef int v4i_vs __attribute__((vector_size(16)));
typedef __attribute__((address_space(1))) v4i_vs* gptr_v4i;
typedef __attribute__((address_space(3))) v4i_vs* lptr_v4i;

#define BM 128
#define BN 64
#define BK 32

// CDNA5 async global->LDS path (ASYNCcnt). Fall back to a register-pipelined
// synchronous copy if this toolchain doesn't declare the builtins.
#if __has_builtin(__builtin_amdgcn_global_load_async_to_lds_b128) && \
    __has_builtin(__builtin_amdgcn_s_wait_asynccnt)
#define USE_ASYNC_LDS 1
#else
#define USE_ASYNC_LDS 0
#endif

__device__ __forceinline__ unsigned short f2bf(float f) {
  union { float f; unsigned u; } v; v.f = f;
  unsigned u = v.u;
  unsigned r = (u + 0x7FFFu + ((u >> 16) & 1u)) >> 16;  // RNE
  return (unsigned short)r;
}

// Load one 16x32 bf16 operand fragment from an [idx][K] row-major LDS tile.
// ISA 16-bit operand layout: lane L holds row/col (L&15), K-half (L>>4);
// each half is two contiguous 8-element (16B) runs -> two ds_load_b128.
__device__ __forceinline__ v16bf load_frag(const unsigned short* lds, int idx, int kh) {
  const v4u* p0 = (const v4u*)(lds + idx * BK + kh * 8);
  const v4u* p1 = (const v4u*)(lds + idx * BK + 16 + kh * 8);
  v4u lo = *p0, hi = *p1;
  v8u r;
  r[0] = lo[0]; r[1] = lo[1]; r[2] = lo[2]; r[3] = lo[3];
  r[4] = hi[0]; r[5] = hi[1]; r[6] = hi[2]; r[7] = hi[3];
  return __builtin_bit_cast(v16bf, r);
}

// ---------------------------------------------------------------------------
// WMMA GEMM: C[M,N] = A[M,K] * B[K,N]; A,B bf16, C f32 accum.
// BT=1: B(k,n) = W[n*ldb + k] (B = W^T, W row-major [N][K]).
// MODE: 0 = write bf16 to Cb, 1 = write f32 to Cf, 2 = Cf += result.
// 256 threads = 8 waves; block tile 128x64; wave tile 16x64; LDS double-buffer.
// ---------------------------------------------------------------------------
template <int BT, int MODE>
__global__ __launch_bounds__(256) void k_wmma_gemm(
    const unsigned short* __restrict__ A, int lda,
    const unsigned short* __restrict__ B, int ldb,
    float* __restrict__ Cf, unsigned short* __restrict__ Cb, int ldc, int K) {
  __shared__ unsigned short Alds[2][BM * BK];   // [m][k], 2 x 8 KB
  __shared__ unsigned short Blds[2][BN * BK];   // [n][k], 2 x 4 KB

  const int tid  = threadIdx.x;
  const int wave = tid >> 5;
  const int lane = tid & 31;
  const int kh   = lane >> 4;
  const int li   = lane & 15;
  const long rowA0 = (long)blockIdx.x * BM;
  const long colB0 = (long)blockIdx.y * BN;

  // A staging: 512 x 16B chunks, 2 per thread (rows tid>>2 and +64).
  const int ar0  = tid >> 2;
  const int ac   = (tid & 3) * 8;
  const unsigned short* aSrc0 = A + (rowA0 + ar0) * (long)lda + ac;
  const unsigned short* aSrc1 = aSrc0 + 64 * (long)lda;
  const int aDst0 = ar0 * BK + ac;
  const int aDst1 = aDst0 + 64 * BK;

  // B staging: one 16B chunk per thread.
  int bR, bC;
  const unsigned short* bSrc;
  if (BT == 0) { bR = tid >> 3; bC = (tid & 7) * 8;           // k, n0
                 bSrc = B + (long)bR * ldb + colB0 + bC; }
  else         { bR = tid >> 2; bC = (tid & 3) * 8;           // n, k0
                 bSrc = B + (long)(colB0 + bR) * ldb + bC; }
  const long bStep = (BT == 0) ? (long)BK * ldb : (long)BK;

  const int nk = K / BK;

  v8f acc[4];
#pragma unroll
  for (int t = 0; t < 4; ++t) acc[t] = (v8f)0.0f;

#if USE_ASYNC_LDS
#define ISSUE_A(buf, koff)                                                     \
  do {                                                                         \
    __builtin_amdgcn_global_load_async_to_lds_b128(                            \
        (gptr_v4i)(aSrc0 + (koff)),                                            \
        (lptr_v4i)(&Alds[buf][aDst0]), 0, 0);                                  \
    __builtin_amdgcn_global_load_async_to_lds_b128(                            \
        (gptr_v4i)(aSrc1 + (koff)),                                            \
        (lptr_v4i)(&Alds[buf][aDst1]), 0, 0);                                  \
  } while (0)
  ISSUE_A(0, 0);                       // prologue: tile 0 -> buffer 0
#else
  v4u aReg0 = *(const v4u*)aSrc0;      // prologue: tile 0 -> registers
  v4u aReg1 = *(const v4u*)aSrc1;
#endif
  v4u bReg = *(const v4u*)bSrc;

  for (int kb = 0; kb < nk; ++kb) {
    const int  cur  = kb & 1;
    const bool more = (kb + 1 < nk);
    unsigned short* Ab = Alds[cur];
    unsigned short* Bb = Blds[cur];

#if USE_ASYNC_LDS
    if (more) ISSUE_A(cur ^ 1, (long)(kb + 1) * BK);  // next tile, other buffer
#else
    *(v4u*)(Ab + aDst0) = aReg0;       // commit registers staged last iter
    *(v4u*)(Ab + aDst1) = aReg1;
#endif
    if (BT == 0) {                     // transpose B chunk into [n][k]
      unsigned short tmp[8];
      *(v4u*)tmp = bReg;
#pragma unroll
      for (int j = 0; j < 8; ++j) Bb[(bC + j) * BK + bR] = tmp[j];
    } else {
      *(v4u*)(Bb + bR * BK + bC) = bReg;
    }
    if (more) {                        // stage next tile into registers
#if !USE_ASYNC_LDS
      aReg0 = *(const v4u*)(aSrc0 + (long)(kb + 1) * BK);
      aReg1 = *(const v4u*)(aSrc1 + (long)(kb + 1) * BK);
#endif
      bReg = *(const v4u*)(bSrc + (long)(kb + 1) * bStep);
    }
#if USE_ASYNC_LDS
    // Async loads complete in order per wave: waiting <=2 (the 2 in flight for
    // the next buffer) guarantees this buffer's copies have landed.
    if (more) __builtin_amdgcn_s_wait_asynccnt(2);
    else      __builtin_amdgcn_s_wait_asynccnt(0);
#endif
    __syncthreads();

    v16bf af = load_frag(Ab, wave * 16 + li, kh);
#pragma unroll
    for (int nt = 0; nt < 4; ++nt) {
      v16bf bf = load_frag(Bb, nt * 16 + li, kh);
      acc[nt] = __builtin_amdgcn_wmma_f32_16x16x32_bf16(
          false, af, false, bf, (short)0, acc[nt], false, false);
    }
    __syncthreads();
  }

  // Epilogue. C/D layout: lane<16 -> (M=v, N=lane); lane>=16 -> (M=v+8, N=lane-16).
#pragma unroll
  for (int nt = 0; nt < 4; ++nt) {
    long col = colB0 + nt * 16 + li;
#pragma unroll
    for (int v = 0; v < 8; ++v) {
      long row = rowA0 + wave * 16 + kh * 8 + v;
      long idx = row * (long)ldc + col;
      float val = acc[nt][v];
      if (MODE == 0)      Cb[idx] = f2bf(val);
      else if (MODE == 1) Cf[idx] = val;
      else                Cf[idx] += val;
    }
  }
}

// ---------------------------------------------------------------------------
__global__ void k_f32_to_bf16(const float* __restrict__ src,
                              unsigned short* __restrict__ dst, long n) {
  long i = (long)blockIdx.x * 256 + threadIdx.x;
  if (i < n) dst[i] = f2bf(src[i]);
}

// Transposed convert: dst[c*R + r] = bf16(src[r*C + c]); 32x32 LDS tiles.
__global__ __launch_bounds__(256) void k_transpose_bf16(
    const float* __restrict__ src, unsigned short* __restrict__ dst,
    int R, int C) {
  __shared__ float tile[32][33];
  const int tx = threadIdx.x & 31, ty = threadIdx.x >> 5;
  const int c0 = blockIdx.x * 32, r0 = blockIdx.y * 32;
#pragma unroll
  for (int i = 0; i < 32; i += 8)
    tile[ty + i][tx] = src[(long)(r0 + ty + i) * C + c0 + tx];
  __syncthreads();
#pragma unroll
  for (int i = 0; i < 32; i += 8)
    dst[(long)(c0 + ty + i) * R + r0 + tx] = f2bf(tile[tx][ty + i]);
}

// Naive f32 linear for the odd-K input projections: out = A @ W^T + b.
__global__ void k_linear_naive(const float* __restrict__ A,
                               const float* __restrict__ W,
                               const float* __restrict__ b,
                               float* __restrict__ outF,
                               unsigned short* __restrict__ outB,
                               int M, int N, int K) {
  long i = (long)blockIdx.x * 256 + threadIdx.x;
  if (i >= (long)M * N) return;
  int r = (int)(i / N), c = (int)(i % N);
  float acc = b[c];
  for (int k = 0; k < K; ++k)
    acc = fmaf(A[(long)r * K + k], W[(long)c * K + k], acc);
  outF[i] = acc;
  outB[i] = f2bf(acc);
}

__global__ void k_add3(const float* a, const float* b, const float* c,
                       float* o, int n) {
  int i = blockIdx.x * 256 + threadIdx.x;
  if (i < n) o[i] = a[i] + b[i] + c[i];
}

// Column mean / biased variance of Z = H + X + bias, over R rows.
__global__ __launch_bounds__(256) void k_colstats(
    const float* __restrict__ H, const float* __restrict__ X,
    const float* __restrict__ bias, float* __restrict__ mean,
    float* __restrict__ var, int R, int C) {
  __shared__ float s1[256], s2[256];
  const int c = blockIdx.x;
  const float bc = bias[c];
  float a = 0.f, q = 0.f;
  for (int r = threadIdx.x; r < R; r += 256) {
    float z = H[(long)r * C + c] + X[(long)r * C + c] + bc;
    a += z;
    q += z * z;
  }
  s1[threadIdx.x] = a; s2[threadIdx.x] = q;
  __syncthreads();
  for (int s = 128; s > 0; s >>= 1) {
    if (threadIdx.x < s) {
      s1[threadIdx.x] += s1[threadIdx.x + s];
      s2[threadIdx.x] += s2[threadIdx.x + s];
    }
    __syncthreads();
  }
  if (threadIdx.x == 0) {
    float m = s1[0] / (float)R;
    mean[c] = m;
    var[c]  = s2[0] / (float)R - m * m;
  }
}

// Xnew = relu(BN(H + X + bias)); write f32 state + bf16 copy for next layer.
__global__ void k_bn_relu(const float* __restrict__ H,
                          const float* __restrict__ X,
                          const float* __restrict__ bias,
                          const float* __restrict__ mean,
                          const float* __restrict__ var,
                          const float* __restrict__ g,
                          const float* __restrict__ be,
                          float* __restrict__ outF,
                          unsigned short* __restrict__ outB,
                          int R, int C) {
  long i = (long)blockIdx.x * 256 + threadIdx.x;
  if (i >= (long)R * C) return;
  int c = (int)(i % C);
  float z = H[i] + X[i] + bias[c];
  float y = (z - mean[c]) * rsqrtf(var[c] + 1e-5f) * g[c] + be[c];
  y = y > 0.f ? y : 0.f;
  outF[i] = y;
  outB[i] = f2bf(y);
}

// Deterministic segment mean: one block per graph, one thread per column.
__global__ __launch_bounds__(128) void k_graph_mean(
    const float* __restrict__ X0, const int* __restrict__ bidx,
    float* __restrict__ out, int R, int C) {
  __shared__ int bb[128];
  const int g = blockIdx.x, c = threadIdx.x;
  float acc = 0.f;
  int cnt = 0;
  for (int r0 = 0; r0 < R; r0 += 128) {
    bb[c] = bidx[r0 + c];
    __syncthreads();
    for (int j = 0; j < 128; ++j) {
      if (bb[j] == g) { acc += X0[(long)(r0 + j) * C + c]; ++cnt; }
    }
    __syncthreads();
  }
  out[(long)g * C + c] = acc / fmaxf((float)cnt, 1.0f);
}

// ============================================================================
// Host orchestration
// ============================================================================
extern "C" void kernel_launch(void* const* d_in, const int* in_sizes, int n_in,
                              void* d_out, int out_size, void* d_ws, size_t ws_size,
                              hipStream_t stream) {
  (void)in_sizes; (void)n_in; (void)out_size; (void)ws_size;

  constexpr int N0 = 4096, N1 = 8192, N2 = 1024;
  constexpr int D0 = 128, D1 = 64, D2 = 128;
  constexpr int DIN0 = 60, DIN1 = 122, DIN2 = 64;
  constexpr int G = 32;
  constexpr int LAYERS = 6;

  const float* x    = (const float*)d_in[0];
  const float* ea   = (const float*)d_in[1];
  const float* sa   = (const float*)d_in[2];
  const float* B0   = (const float*)d_in[3];
  const float* B1   = (const float*)d_in[4];
  const float* A1   = (const float*)d_in[5];
  const int*   bidx = (const int*)d_in[6];
  const float* Win0W = (const float*)d_in[7];   const float* Win0b = (const float*)d_in[8];
  const float* Win1W = (const float*)d_in[9];   const float* Win1b = (const float*)d_in[10];
  const float* Win2W = (const float*)d_in[11];  const float* Win2b = (const float*)d_in[12];
  auto WE  = [&](int fr, int to) { return (const float*)d_in[13 + 2 * (fr * 3 + to)]; };
  auto WEb = [&](int fr, int to) { return (const float*)d_in[14 + 2 * (fr * 3 + to)]; };
  const float* g0 = (const float*)d_in[31]; const float* be0 = (const float*)d_in[32];
  const float* g1 = (const float*)d_in[33]; const float* be1 = (const float*)d_in[34];
  const float* g2 = (const float*)d_in[35]; const float* be2 = (const float*)d_in[36];

  // ---- workspace carve-up (256B aligned) ----------------------------------
  size_t off = 0;
  auto carve = [&](size_t bytes) -> char* {
    char* p = (char*)d_ws + off;
    off = (off + bytes + 255) & ~(size_t)255;
    return p;
  };
  unsigned short* B0b  = (unsigned short*)carve((size_t)N0 * N1 * 2);
  unsigned short* B0Tb = (unsigned short*)carve((size_t)N1 * N0 * 2);
  unsigned short* A1Tb = (unsigned short*)carve((size_t)N1 * N1 * 2);
  unsigned short* B1b  = (unsigned short*)carve((size_t)N1 * N2 * 2);
  unsigned short* B1Tb = (unsigned short*)carve((size_t)N2 * N1 * 2);
  unsigned short* W10b = (unsigned short*)carve((size_t)D0 * D1 * 2);
  unsigned short* W01b = (unsigned short*)carve((size_t)D1 * D0 * 2);
  unsigned short* W11b = (unsigned short*)carve((size_t)D1 * D1 * 2);
  unsigned short* W21b = (unsigned short*)carve((size_t)D1 * D2 * 2);
  unsigned short* W12b = (unsigned short*)carve((size_t)D2 * D1 * 2);
  float* bias0 = (float*)carve(D0 * 4);
  float* bias1 = (float*)carve(D1 * 4);
  float* bias2 = (float*)carve(D2 * 4);
  float* X0f = (float*)carve((size_t)N0 * D0 * 4);
  float* X1f = (float*)carve((size_t)N1 * D1 * 4);
  float* X2f = (float*)carve((size_t)N2 * D2 * 4);
  unsigned short* X0b = (unsigned short*)carve((size_t)N0 * D0 * 2);
  unsigned short* X1b = (unsigned short*)carve((size_t)N1 * D1 * 2);
  unsigned short* X2b = (unsigned short*)carve((size_t)N2 * D2 * 2);
  unsigned short* T0b  = (unsigned short*)carve((size_t)N0 * D1 * 2);
  unsigned short* T1ab = (unsigned short*)carve((size_t)N1 * D0 * 2);
  unsigned short* T1bb = (unsigned short*)carve((size_t)N1 * D1 * 2);
  unsigned short* T1cb = (unsigned short*)carve((size_t)N1 * D0 * 2);
  unsigned short* T2b  = (unsigned short*)carve((size_t)N2 * D1 * 2);
  float* H0 = (float*)carve((size_t)N0 * D0 * 4);
  float* H1 = (float*)carve((size_t)N1 * D1 * 4);
  float* H2 = (float*)carve((size_t)N2 * D2 * 4);
  float* mean0 = (float*)carve(D0 * 4); float* var0 = (float*)carve(D0 * 4);
  float* mean1 = (float*)carve(D1 * 4); float* var1 = (float*)carve(D1 * 4);
  float* mean2 = (float*)carve(D2 * 4); float* var2 = (float*)carve(D2 * 4);

  float* outX0 = (float*)d_out;
  float* outX1 = outX0 + (size_t)N0 * D0;
  float* outX2 = outX1 + (size_t)N1 * D1;
  float* outG  = outX2 + (size_t)N2 * D2;

  auto cvt = [&](const float* s, unsigned short* d, long n) {
    k_f32_to_bf16<<<(unsigned)((n + 255) / 256), 256, 0, stream>>>(s, d, n);
  };

  // ---- one-time bf16 conversions / transposes -----------------------------
  cvt(B0, B0b, (long)N0 * N1);
  k_transpose_bf16<<<dim3(N1 / 32, N0 / 32), 256, 0, stream>>>(B0, B0Tb, N0, N1);
  cvt(B1, B1b, (long)N1 * N2);
  k_transpose_bf16<<<dim3(N2 / 32, N1 / 32), 256, 0, stream>>>(B1, B1Tb, N1, N2);
  k_transpose_bf16<<<dim3(N1 / 32, N1 / 32), 256, 0, stream>>>(A1, A1Tb, N1, N1);
  cvt(WE(1, 0), W10b, (long)D0 * D1);
  cvt(WE(0, 1), W01b, (long)D1 * D0);
  cvt(WE(1, 1), W11b, (long)D1 * D1);
  cvt(WE(2, 1), W21b, (long)D1 * D2);
  cvt(WE(1, 2), W12b, (long)D2 * D1);
  k_add3<<<1, 256, 0, stream>>>(WEb(0, 0), WEb(1, 0), WEb(2, 0), bias0, D0);
  k_add3<<<1, 256, 0, stream>>>(WEb(0, 1), WEb(1, 1), WEb(2, 1), bias1, D1);
  k_add3<<<1, 256, 0, stream>>>(WEb(0, 2), WEb(1, 2), WEb(2, 2), bias2, D2);

  // ---- input projections (K = 60/122/64; tiny, scalar f32) ----------------
  k_linear_naive<<<(N0 * D0 + 255) / 256, 256, 0, stream>>>(x,  Win0W, Win0b, X0f, X0b, N0, D0, DIN0);
  k_linear_naive<<<(N1 * D1 + 255) / 256, 256, 0, stream>>>(ea, Win1W, Win1b, X1f, X1b, N1, D1, DIN1);
  k_linear_naive<<<(N2 * D2 + 255) / 256, 256, 0, stream>>>(sa, Win2W, Win2b, X2f, X2b, N2, D2, DIN2);

  // Template-dispatched GEMM launchers.
  auto gemm_agg = [&](const unsigned short* A, int lda, const unsigned short* B,
                      int ldb, unsigned short* Cb, int ldc, int M, int N, int K) {
    k_wmma_gemm<0, 0><<<dim3(M / BM, N / BN), 256, 0, stream>>>(
        A, lda, B, ldb, nullptr, Cb, ldc, K);
  };
  auto gemm_lin = [&](const unsigned short* A, int lda, const unsigned short* W,
                      int ldw, float* Cf, int ldc, int M, int N, int K, bool accum) {
    if (accum)
      k_wmma_gemm<1, 2><<<dim3(M / BM, N / BN), 256, 0, stream>>>(
          A, lda, W, ldw, Cf, nullptr, ldc, K);
    else
      k_wmma_gemm<1, 1><<<dim3(M / BM, N / BN), 256, 0, stream>>>(
          A, lda, W, ldw, Cf, nullptr, ldc, K);
  };

  // ---- 6 message-passing layers -------------------------------------------
  for (int L = 0; L < LAYERS; ++L) {
    // aggregation (WMMA bf16 -> bf16)
    gemm_agg(B0b,  N1, X1b, D1, T0b,  D1, N0, D1, N1);  // B0 @ X1
    gemm_agg(B0Tb, N0, X0b, D0, T1ab, D0, N1, D0, N0);  // B0^T @ X0
    gemm_agg(A1Tb, N1, X1b, D1, T1bb, D1, N1, D1, N1);  // A1^T @ X1
    gemm_agg(B1b,  N2, X2b, D2, T1cb, D2, N1, D2, N2);  // B1 @ X2
    gemm_agg(B1Tb, N1, X1b, D1, T2b,  D1, N2, D1, N1);  // B1^T @ X1
    // per-type linears (WMMA bf16 -> f32, accumulate the three H1 terms)
    gemm_lin(T0b,  D1, W10b, D1, H0, D0, N0, D0, D1, false);
    gemm_lin(T1ab, D0, W01b, D0, H1, D1, N1, D1, D0, false);
    gemm_lin(T1bb, D1, W11b, D1, H1, D1, N1, D1, D1, true);
    gemm_lin(T1cb, D0, W21b, D0, H1, D1, N1, D1, D0, true);
    gemm_lin(T2b,  D1, W12b, D1, H2, D2, N2, D2, D1, false);
    // residual + BN + ReLU
    const bool last = (L == LAYERS - 1);
    k_colstats<<<D0, 256, 0, stream>>>(H0, X0f, bias0, mean0, var0, N0, D0);
    k_bn_relu<<<(N0 * D0 + 255) / 256, 256, 0, stream>>>(
        H0, X0f, bias0, mean0, var0, g0, be0, last ? outX0 : X0f, X0b, N0, D0);
    k_colstats<<<D1, 256, 0, stream>>>(H1, X1f, bias1, mean1, var1, N1, D1);
    k_bn_relu<<<(N1 * D1 + 255) / 256, 256, 0, stream>>>(
        H1, X1f, bias1, mean1, var1, g1, be1, last ? outX1 : X1f, X1b, N1, D1);
    k_colstats<<<D2, 256, 0, stream>>>(H2, X2f, bias2, mean2, var2, N2, D2);
    k_bn_relu<<<(N2 * D2 + 255) / 256, 256, 0, stream>>>(
        H2, X2f, bias2, mean2, var2, g2, be2, last ? outX2 : X2f, X2b, N2, D2);
  }

  // ---- graph readout ------------------------------------------------------
  k_graph_mean<<<G, 128, 0, stream>>>(outX0, bidx, outG, N0, D0);
}
